// MScaleAddTreeEnergyLoss_51161650430625
// MI455X (gfx1250) — matrix operand order repaired and checked by reading, not compiled, not run on hardware
//
#include <hip/hip_runtime.h>
#include <hip/hip_bf16.h>
#include <stdint.h>

// MScaleAddTreeEnergyLoss for MI455X (gfx1250).
// B=4, C=4, H=W=96, N=9216, E=18240 edges per grid graph.
// Strategy: latency-bound graph workload -> keep per-tree sort (256KB) and the
// full scan state (184KB) in CDNA5's 320KB LDS; map the only dense-math part
// (multi-channel edge weights) onto v_wmma_f32_16x16x32_bf16 gram diagonals.

#define SIGMA_F 0.02f

typedef __attribute__((ext_vector_type(16))) __bf16 v16bf;
typedef __attribute__((ext_vector_type(8)))  float  v8f;

static __device__ __forceinline__ __bf16 f2bf(float f) {
  union { float f; uint32_t u; } x; x.f = f;
  uint32_t r = x.u + 0x7FFFu + ((x.u >> 16) & 1u);   // round-to-nearest-even
  uint16_t hb = (uint16_t)(r >> 16);
  __bf16 h; __builtin_memcpy(&h, &hb, 2);
  return h;
}

// ---------------------------------------------------------------- upsample
__global__ void k_upsample(const float* __restrict__ src, float* __restrict__ dst,
                           int B_, int Cc, int sh, int sw, int H_, int W_) {
  int i = blockIdx.x * blockDim.x + threadIdx.x;
  int tot = B_ * Cc * H_ * W_;
  if (i >= tot) return;
  int x = i % W_;
  int y = (i / W_) % H_;
  int c = (i / (W_ * H_)) % Cc;
  int b = i / (W_ * H_ * Cc);
  float sy = ((y + 0.5f) * sh) / H_ - 0.5f;
  float sx = ((x + 0.5f) * sw) / W_ - 0.5f;
  int y0 = (int)floorf(sy), x0 = (int)floorf(sx);
  float fy = sy - (float)y0, fx = sx - (float)x0;
  int y0c = y0 < 0 ? 0 : (y0 > sh - 1 ? sh - 1 : y0);
  int y1 = y0 + 1; int y1c = y1 < 0 ? 0 : (y1 > sh - 1 ? sh - 1 : y1);
  int x0c = x0 < 0 ? 0 : (x0 > sw - 1 ? sw - 1 : x0);
  int x1 = x0 + 1; int x1c = x1 < 0 ? 0 : (x1 > sw - 1 ? sw - 1 : x1);
  const float* s = src + ((size_t)b * Cc + c) * (size_t)sh * sw;
  float v00 = s[y0c * sw + x0c], v01 = s[y0c * sw + x1c];
  float v10 = s[y1c * sw + x0c], v11 = s[y1c * sw + x1c];
  float v0 = v00 + fx * (v01 - v00);
  float v1 = v10 + fx * (v11 - v10);
  dst[i] = v0 + fy * (v1 - v0);
}

// ---------------------------------------------------------------- softmax C=4
__global__ void k_softmax(const float* __restrict__ p, float* __restrict__ out,
                          int B_, int N_) {
  int i = blockIdx.x * blockDim.x + threadIdx.x;
  if (i >= B_ * N_) return;
  int b = i / N_, v = i % N_;
  size_t base = (size_t)b * 4 * N_ + v;
  float a0 = p[base], a1 = p[base + N_], a2 = p[base + 2 * (size_t)N_], a3 = p[base + 3 * (size_t)N_];
  float m = fmaxf(fmaxf(a0, a1), fmaxf(a2, a3));
  float e0 = __expf(a0 - m), e1 = __expf(a1 - m), e2 = __expf(a2 - m), e3 = __expf(a3 - m);
  float inv = 1.0f / (e0 + e1 + e2 + e3);
  out[base] = e0 * inv; out[base + N_] = e1 * inv;
  out[base + 2 * (size_t)N_] = e2 * inv; out[base + 3 * (size_t)N_] = e3 * inv;
}

// ---------------------------------------------------------------- per-pixel ||f||^2
__global__ void k_ssq(const float* __restrict__ f, float* __restrict__ out,
                      int B_, int Cc, int N_) {
  int i = blockIdx.x * blockDim.x + threadIdx.x;
  if (i >= B_ * N_) return;
  int b = i / N_, v = i % N_;
  float s = 0.0f;
  for (int c = 0; c < Cc; ++c) {
    const float* p = f + ((size_t)b * Cc + c) * N_ + v;
    __builtin_prefetch(p + (size_t)4 * N_, 0, 1);
    float x = *p;
    s = fmaf(x, x, s);
  }
  out[(size_t)b * N_ + v] = s;
}

// ---------------------------------------------------------------- low-guide (C=3) edge weights
__global__ void k_edge_low(const float* __restrict__ f, float* __restrict__ wsq,
                           int B_, int N_) {
  const int E_ = 18240, VE = 9120;
  int i = blockIdx.x * blockDim.x + threadIdx.x;
  if (i >= B_ * E_) return;
  int b = i / E_, e = i % E_;
  int u, v;
  if (e < VE) { int y = e / 96, x = e % 96; u = y * 96 + x; v = u + 96; }
  else        { int eh = e - VE; int y = eh / 95, x = eh % 95; u = y * 96 + x; v = u + 1; }
  float s = 0.0f;
  for (int c = 0; c < 3; ++c) {
    float d = f[((size_t)b * 3 + c) * N_ + u] - f[((size_t)b * 3 + c) * N_ + v];
    s = fmaf(d, d, s);
  }
  wsq[(size_t)b * E_ + e] = s;
}

// ---------------------------------------------------------------- WMMA edge weights
// One wave per 16-edge tile; dist^2 = ssq[u] + ssq[v] - 2*dot(u,v), dot via
// the diagonal of a 16x16 bf16 WMMA (A = 16 pixels x 32ch, B = shifted pixels).
__global__ void k_edge_wmma(const float* __restrict__ feat, const float* __restrict__ ssq,
                            float* __restrict__ wsq, int B_, int Cc, int N_) {
  const int E_ = 18240, VE = 9120;
  const int VT = 95 * 6, TILES = VT + 96 * 6;   // vertical tiles, then horizontal
  int wave = (blockIdx.x * blockDim.x + threadIdx.x) >> 5;
  int lane = threadIdx.x & 31;
  if (wave >= B_ * TILES) return;
  int b = wave / TILES;
  int t = wave % TILES;
  bool isVert = (t < VT);
  int y, xb;
  if (isVert) { y = t / 6; xb = t % 6; }
  else        { int th = t - VT; y = th / 6; xb = th % 6; }
  int x0 = xb * 16;
  int half16 = lane >> 4;
  int ml = lane & 15;
  int pixA = x0 + ml, rowA = y;
  int pixB, rowB;
  if (isVert) { pixB = pixA; rowB = y + 1; }
  else        { pixB = (pixA + 1 > 95) ? 95 : pixA + 1; rowB = y; }
  size_t baseA = (size_t)rowA * 96 + pixA;
  size_t baseB = (size_t)rowB * 96 + pixB;
  const float* fb = feat + (size_t)b * Cc * N_;

  v8f acc = {};
  int chunks = Cc >> 5;
  for (int kc = 0; kc < chunks; ++kc) {
    v16bf av, bv;
#pragma unroll
    for (int h = 0; h < 16; ++h) {
      // A-matrix 16-bit layout: lanes<16 hold K 0..7,16..23; lanes>=16 add 8.
      int j = h >> 1, p0 = h & 1;
      int ka = ((j < 4) ? (2 * j + p0) : (16 + 2 * (j - 4) + p0)) + (half16 ? 8 : 0);
      // B-matrix 32x16: same K order as A (lanes<16 hold K 0..7,16..23; lanes>=16 add 8).
      int kb = ((h < 8) ? h : h + 8) + (half16 ? 8 : 0);
      av[h] = f2bf(fb[(size_t)(kc * 32 + ka) * N_ + baseA]);
      bv[h] = f2bf(fb[(size_t)(kc * 32 + kb) * N_ + baseB]);
    }
    acc = __builtin_amdgcn_wmma_f32_16x16x32_bf16(false, av, false, bv,
                                                  (short)0, acc, false, false);
  }

  // Diagonal D[m][m]: VGPR r at lane r (m=r, r<8) and lane 24+r (m=8+r).
  int m = -1;
  if (lane < 8) m = lane;
  else if (lane >= 24) m = lane - 16;
  float dot = 0.0f;
#pragma unroll
  for (int r = 0; r < 8; ++r)
    if ((lane < 8 && lane == r) || (lane >= 24 && lane - 24 == r)) dot = acc[r];

  if (m >= 0) {
    int x = x0 + m;
    const float* sb = ssq + (size_t)b * N_;
    if (isVert) {
      int u = y * 96 + x, v = u + 96;
      wsq[(size_t)b * E_ + y * 96 + x] = sb[u] + sb[v] - 2.0f * dot;
    } else if (x <= 94) {
      int u = y * 96 + x, v = u + 1;
      wsq[(size_t)b * E_ + VE + y * 95 + x] = sb[u] + sb[v] - 2.0f * dot;
    }
  }
}

// ---------------------------------------------------------------- tree build
// One workgroup per tree. 32K-key bitonic sort in 256KB LDS (CDNA5-only),
// then serial Kruskal + BFS by thread 0 in global scratch.
__global__ void __launch_bounds__(1024)
k_build_tree(const float* __restrict__ wsq, int* __restrict__ scratch,
             int* __restrict__ parentOut, int* __restrict__ orderOut, int N_) {
  const int E_ = 18240, VE = 9120, NS = 32768;
  extern __shared__ unsigned long long sk[];
  int b = blockIdx.x;
  const float* w = wsq + (size_t)b * E_;
  int* sbase = scratch + (size_t)b * 10 * N_;
  int* uf    = sbase;
  int* eu    = sbase + N_;
  int* ev    = sbase + 2 * N_;
  int* deg   = sbase + 3 * N_;
  int* start = sbase + 4 * N_;            // N_+1 entries
  int* pos   = sbase + 5 * N_ + 16;
  int* adj   = sbase + 6 * N_;            // 2*(N_-1) entries
  int* pOut  = parentOut + (size_t)b * N_;
  int* oOut  = orderOut + (size_t)b * N_;
  int tid = threadIdx.x;

  for (int i = tid; i < NS; i += 1024) {
    unsigned long long key;
    if (i < E_) key = ((unsigned long long)__float_as_uint(w[i]) << 32) | (unsigned)i;
    else        key = ~0ULL;
    sk[i] = key;
  }
  for (int i = tid; i < N_; i += 1024) { uf[i] = i; deg[i] = 0; pOut[i] = -1; }
  __syncthreads();

  // bitonic sort (stable on ties via index in low bits, matching stable argsort)
  for (int k = 2; k <= NS; k <<= 1) {
    for (int j = k >> 1; j > 0; j >>= 1) {
      for (int t = tid; t < NS; t += 1024) {
        int ixj = t ^ j;
        if (ixj > t) {
          unsigned long long a = sk[t], c = sk[ixj];
          bool up = ((t & k) == 0);
          if ((a > c) == up) { sk[t] = c; sk[ixj] = a; }
        }
      }
      __syncthreads();
    }
  }

  if (tid == 0) {
    // Kruskal
    int cnt = 0;
    for (int s = 0; s < NS && cnt < N_ - 1; ++s) {
      int e = (int)(sk[s] & 0xFFFFFFFFu);
      if (e >= E_) continue;
      int u, v;
      if (e < VE) { int y = e / 96, x = e % 96; u = y * 96 + x; v = u + 96; }
      else        { int eh = e - VE; int y = eh / 95, x = eh % 95; u = y * 96 + x; v = u + 1; }
      int ru = u; while (uf[ru] != ru) { uf[ru] = uf[uf[ru]]; ru = uf[ru]; }
      int rv = v; while (uf[rv] != rv) { uf[rv] = uf[uf[rv]]; rv = uf[rv]; }
      if (ru != rv) { uf[ru] = rv; eu[cnt] = u; ev[cnt] = v; ++cnt; }
    }
    // adjacency
    for (int k2 = 0; k2 < cnt; ++k2) { deg[eu[k2]]++; deg[ev[k2]]++; }
    start[0] = 0;
    for (int i = 0; i < N_; ++i) start[i + 1] = start[i] + deg[i];
    for (int i = 0; i < N_; ++i) pos[i] = start[i];
    for (int k2 = 0; k2 < cnt; ++k2) {
      adj[pos[eu[k2]]++] = ev[k2];
      adj[pos[ev[k2]]++] = eu[k2];
    }
    // BFS from 0
    pOut[0] = 0; oOut[0] = 0;
    int head = 0, tail = 1;
    while (head < tail) {
      int xv = oOut[head++];
      for (int jj = start[xv]; jj < start[xv + 1]; ++jj) {
        int yv = adj[jj];
        if (pOut[yv] < 0) { pOut[yv] = xv; oOut[tail++] = yv; }
      }
    }
  }
}

// ---------------------------------------------------------------- edge weights w[v] = exp(-d2/sigma)
__global__ void k_compw(const float* __restrict__ embed, const int* __restrict__ parent,
                        float* __restrict__ wv, int B_, int Cc, int N_) {
  int i = blockIdx.x * blockDim.x + threadIdx.x;
  if (i >= B_ * N_) return;
  int b = i / N_, v = i % N_;
  int p = parent[(size_t)b * N_ + v];
  float s = 0.0f;
  for (int c = 0; c < Cc; ++c) {
    const float* e = embed + ((size_t)b * Cc + c) * N_;
    float d = e[v] - e[p];
    s = fmaf(d, d, s);
  }
  wv[(size_t)b * N_ + v] = __expf(-s / SIGMA_F);
}

// ---------------------------------------------------------------- normalized tree filter
// One workgroup per tree; A[N][5] (feat channels + normalizer) resident in LDS.
// Up pass (reverse BFS) and in-place down pass done by one wave, 5 lanes = 5 ch.
__global__ void __launch_bounds__(256)
k_tree_filter(const float* __restrict__ feat, const int* __restrict__ parent,
              const int* __restrict__ order, const float* __restrict__ wv,
              float* __restrict__ out, int N_) {
  extern __shared__ float A[];
  int b = blockIdx.x;
  int tid = threadIdx.x;
  const float* fb = feat + (size_t)b * 4 * N_;
  const int* par = parent + (size_t)b * N_;
  const int* ord = order + (size_t)b * N_;
  const float* w = wv + (size_t)b * N_;
  for (int i = tid; i < N_; i += 256) {
    __builtin_prefetch(fb + i + 256, 0, 1);
    A[i * 5 + 0] = fb[i];
    A[i * 5 + 1] = fb[N_ + i];
    A[i * 5 + 2] = fb[2 * (size_t)N_ + i];
    A[i * 5 + 3] = fb[3 * (size_t)N_ + i];
    A[i * 5 + 4] = 1.0f;
  }
  __syncthreads();
  if (tid < 32) {
    int lane = tid;
    // leaf -> root aggregation
    for (int i = N_ - 1; i >= 1; --i) {
      int vtx = ord[i]; int p = par[vtx]; float ww = w[vtx];
      if (lane < 5) A[p * 5 + lane] += ww * A[vtx * 5 + lane];
    }
    // root -> leaf distribution (in place: slot[p] already holds S[p])
    for (int i = 1; i < N_; ++i) {
      int vtx = ord[i]; int p = par[vtx]; float ww = w[vtx];
      if (lane < 5) {
        float a = A[vtx * 5 + lane];
        A[vtx * 5 + lane] = a + ww * (A[p * 5 + lane] - ww * a);
      }
    }
  }
  __syncthreads();
  float* ob = out + (size_t)b * 4 * N_;
  for (int i = tid; i < N_; i += 256) {
    float inv = 1.0f / A[i * 5 + 4];
    ob[i]                    = A[i * 5 + 0] * inv;
    ob[N_ + i]               = A[i * 5 + 1] * inv;
    ob[2 * (size_t)N_ + i]   = A[i * 5 + 2] * inv;
    ob[3 * (size_t)N_ + i]   = A[i * 5 + 3] * inv;
  }
}

// ---------------------------------------------------------------- masked L1 reduction
__global__ void k_loss(const float* __restrict__ prob, const float* __restrict__ a1,
                       const float* __restrict__ a2, const float* __restrict__ a3,
                       const float* __restrict__ roi, float* __restrict__ accum,
                       int B_, int N_) {
  int i = blockIdx.x * blockDim.x + threadIdx.x;
  if (i >= B_ * N_) return;
  int b = i / N_, v = i % N_;
  float r = roi[(size_t)b * N_ + v];
  float l = 0.0f;
  for (int c = 0; c < 4; ++c) {
    size_t idx = ((size_t)b * 4 + c) * N_ + v;
    float p = prob[idx];
    l += fabsf(p - a1[idx]) + fabsf(p - a2[idx]) + fabsf(p - a3[idx]);
  }
  l *= r;
  for (int o = 16; o > 0; o >>= 1) {     // wave32 tree reduce
    l += __shfl_down(l, o, 32);
    r += __shfl_down(r, o, 32);
  }
  if ((threadIdx.x & 31) == 0) {
    atomicAdd(&accum[0], l);
    atomicAdd(&accum[1], r);
  }
}

__global__ void k_init(float* accum) { accum[0] = 0.0f; accum[1] = 0.0f; }

__global__ void k_final(const float* __restrict__ accum, const float* __restrict__ weight,
                        float* __restrict__ out) {
  float L = accum[0], Nn = accum[1];
  out[0] = weight[0] * (Nn > 0.0f ? L / Nn : L);
}

// ---------------------------------------------------------------- host launcher
extern "C" void kernel_launch(void* const* d_in, const int* in_sizes, int n_in,
                              void* d_out, int out_size, void* d_ws, size_t ws_size,
                              hipStream_t stream) {
  (void)in_sizes; (void)n_in; (void)out_size; (void)ws_size;
  const int B = 4, C = 4, H = 96, W = 96, N = H * W, E = 18240;

  const float* preds  = (const float*)d_in[0];
  const float* low    = (const float*)d_in[1];
  const float* hf1    = (const float*)d_in[2];
  const float* hf2    = (const float*)d_in[3];
  const float* hf3    = (const float*)d_in[4];
  const float* roi    = (const float*)d_in[5];
  const float* weight = (const float*)d_in[6];
  float* out = (float*)d_out;

  char* ws = (char*)d_ws;
  size_t off = 0;
  auto take = [&](size_t bytes) -> char* {
    char* p = ws + off;
    off += (bytes + 255) & ~(size_t)255;
    return p;
  };
  float* hf1u  = (float*)take((size_t)B * 64 * N * 4);
  float* hf2u  = (float*)take((size_t)B * 128 * N * 4);
  float* hf3u  = (float*)take((size_t)B * 256 * N * 4);
  float* prob  = (float*)take((size_t)B * 4 * N * 4);
  float* AS    = (float*)take((size_t)B * 4 * N * 4);
  float* ssq   = (float*)take((size_t)3 * B * N * 4);
  float* wsq   = (float*)take((size_t)4 * B * E * 4);
  int*   parent= (int*)  take((size_t)4 * B * N * 4);
  int*   order = (int*)  take((size_t)4 * B * N * 4);
  float* wv    = (float*)take((size_t)4 * B * N * 4);
  int*   tscr  = (int*)  take((size_t)4 * B * 10 * N * 4);
  float* accum = (float*)take(256);

  k_init<<<1, 1, 0, stream>>>(accum);

  { int tot = B * 64 * N;  k_upsample<<<(tot + 255) / 256, 256, 0, stream>>>(hf1, hf1u, B, 64, 48, 48, H, W); }
  { int tot = B * 128 * N; k_upsample<<<(tot + 255) / 256, 256, 0, stream>>>(hf2, hf2u, B, 128, 24, 24, H, W); }
  { int tot = B * 256 * N; k_upsample<<<(tot + 255) / 256, 256, 0, stream>>>(hf3, hf3u, B, 256, 12, 12, H, W); }

  k_softmax<<<(B * N + 255) / 256, 256, 0, stream>>>(preds, prob, B, N);

  // edge weights: scalar for 3-ch low guide, WMMA for the wide guides
  k_edge_low<<<(B * E + 255) / 256, 256, 0, stream>>>(low, wsq, B, N);
  k_ssq<<<(B * N + 255) / 256, 256, 0, stream>>>(hf1u, ssq,             B, 64,  N);
  k_ssq<<<(B * N + 255) / 256, 256, 0, stream>>>(hf2u, ssq + B * N,     B, 128, N);
  k_ssq<<<(B * N + 255) / 256, 256, 0, stream>>>(hf3u, ssq + 2 * B * N, B, 256, N);
  const int TILES = 95 * 6 + 96 * 6;                 // 1146 tiles/batch
  int waves = B * TILES;
  int blks = (waves + 3) / 4;                        // 4 wave32 per block of 128
  k_edge_wmma<<<blks, 128, 0, stream>>>(hf1u, ssq,             wsq + (size_t)1 * B * E, B, 64,  N);
  k_edge_wmma<<<blks, 128, 0, stream>>>(hf2u, ssq + B * N,     wsq + (size_t)2 * B * E, B, 128, N);
  k_edge_wmma<<<blks, 128, 0, stream>>>(hf3u, ssq + 2 * B * N, wsq + (size_t)3 * B * E, B, 256, N);

  // MST + BFS per tree (4 guides x 4 batches); 256KB LDS bitonic sort
  const size_t sortLds = (size_t)32768 * 8;
  for (int g = 0; g < 4; ++g) {
    k_build_tree<<<B, 1024, sortLds, stream>>>(wsq + (size_t)g * B * E,
                                               tscr + (size_t)g * B * 10 * N,
                                               parent + (size_t)g * B * N,
                                               order + (size_t)g * B * N, N);
  }

  k_compw<<<(B * N + 255) / 256, 256, 0, stream>>>(low,  parent,                    wv,                    B, 3,   N);
  k_compw<<<(B * N + 255) / 256, 256, 0, stream>>>(hf1u, parent + (size_t)1 * B * N, wv + (size_t)1 * B * N, B, 64,  N);
  k_compw<<<(B * N + 255) / 256, 256, 0, stream>>>(hf2u, parent + (size_t)2 * B * N, wv + (size_t)2 * B * N, B, 128, N);
  k_compw<<<(B * N + 255) / 256, 256, 0, stream>>>(hf3u, parent + (size_t)3 * B * N, wv + (size_t)3 * B * N, B, 256, N);

  // tree filters: phase A (prob x low-tree -> AS), phase B (AS x hf-trees -> out)
  const size_t filtLds = (size_t)N * 5 * 4;          // 184320 bytes
  k_tree_filter<<<B, 256, filtLds, stream>>>(prob, parent, order, wv, AS, N);
  float* AS1 = out + 1;
  float* AS2 = AS1 + (size_t)B * C * N;
  float* AS3 = AS2 + (size_t)B * C * N;
  k_tree_filter<<<B, 256, filtLds, stream>>>(AS, parent + (size_t)1 * B * N, order + (size_t)1 * B * N,
                                             wv + (size_t)1 * B * N, AS1, N);
  k_tree_filter<<<B, 256, filtLds, stream>>>(AS, parent + (size_t)2 * B * N, order + (size_t)2 * B * N,
                                             wv + (size_t)2 * B * N, AS2, N);
  k_tree_filter<<<B, 256, filtLds, stream>>>(AS, parent + (size_t)3 * B * N, order + (size_t)3 * B * N,
                                             wv + (size_t)3 * B * N, AS3, N);

  k_loss<<<(B * N + 255) / 256, 256, 0, stream>>>(prob, AS1, AS2, AS3, roi, accum, B, N);
  k_final<<<1, 1, 0, stream>>>(accum, weight, out);
}
